// MultiHeadAttention_50268297232472
// MI455X (gfx1250) — matrix operand order repaired and checked by reading, not compiled, not run on hardware
//
#include <hip/hip_runtime.h>
#include <stdint.h>

#define BB 8
#define SS 2048
#define DD 1024
#define HH 16
#define DKK 64

typedef __attribute__((ext_vector_type(16))) __bf16 bf16x16;
typedef __attribute__((ext_vector_type(8)))  float  f32x8;
typedef __attribute__((ext_vector_type(4)))  float  f32x4;
typedef __attribute__((ext_vector_type(4)))  unsigned int u32x4;
typedef __attribute__((ext_vector_type(8)))  int i32x8;
typedef __attribute__((ext_vector_type(4)))  int i32x4;

union Frag  { bf16x16 v; u32x4 q[2]; };
union Pack8 { __bf16 h[8]; u32x4 q; };

__device__ inline __bf16 f2bf(float f) { return (__bf16)f; }

__device__ inline f32x8 zero8() {
  f32x8 z;
#pragma unroll
  for (int i = 0; i < 8; ++i) z[i] = 0.0f;
  return z;
}

__device__ inline f32x8 wmma_bf16(const Frag& a, const Frag& b, f32x8 c) {
  return __builtin_amdgcn_wmma_f32_16x16x32_bf16(false, a.v, false, b.v,
                                                 (short)0, c, false, false);
}

// ---- TDM: 2-D tile load global -> LDS (descriptor per CDNA5 ISA ch.8) ----
__device__ inline void tdm_load_2d(unsigned lds_off, const void* gptr,
                                   unsigned dim0, unsigned dim1,
                                   unsigned long long stride0,
                                   unsigned tile0, unsigned tile1) {
  unsigned long long ga = (unsigned long long)(uintptr_t)gptr;
  u32x4 g0;
  g0[0] = 1u;                                   // count=1, user descriptor
  g0[1] = lds_off;                              // lds_addr (bytes)
  g0[2] = (unsigned)(ga & 0xFFFFFFFFu);         // global_addr lo
  g0[3] = (unsigned)((ga >> 32) & 0x01FFFFFFu) | (2u << 30);  // hi + type=2
  i32x8 g1;
  g1[0] = (int)(1u << 16);                      // data_size=1 -> 2 bytes
  g1[1] = (int)((dim0 & 0xFFFFu) << 16);        // tensor_dim0[15:0] @bits63:48
  g1[2] = (int)(((dim0 >> 16) & 0xFFFFu) | ((dim1 & 0xFFFFu) << 16));
  g1[3] = (int)(((dim1 >> 16) & 0xFFFFu) | ((tile0 & 0xFFFFu) << 16));
  g1[4] = (int)(tile1 & 0xFFFFu);               // tile_dim1, tile_dim2=0
  g1[5] = (int)(unsigned)(stride0 & 0xFFFFFFFFull);
  g1[6] = (int)(unsigned)((stride0 >> 32) & 0xFFFFull);
  g1[7] = 0;
  i32x4 z4; z4[0] = 0; z4[1] = 0; z4[2] = 0; z4[3] = 0;
  i32x8 z8;
#pragma unroll
  for (int i = 0; i < 8; ++i) z8[i] = 0;
  __builtin_amdgcn_tensor_load_to_lds(g0, g1, z4, z4, z8, 0);
}

// ---------------- f32 -> bf16 convert (plain) ----------------
__global__ void cvt_f32_bf16(const float* __restrict__ src,
                             __bf16* __restrict__ dst, int n) {
  int i = blockIdx.x * blockDim.x + threadIdx.x;
  int stride = gridDim.x * blockDim.x;
  for (; i < n; i += stride) dst[i] = f2bf(src[i]);
}

// ---- W[h][d][n] f32 -> Wt[h][n][d] bf16 (pre-transpose for GEMM B-frags) --
__global__ void cvt_transpose_w(const float* __restrict__ W,
                                __bf16* __restrict__ Wt, int n) {
  int i = blockIdx.x * blockDim.x + threadIdx.x;
  int stride = gridDim.x * blockDim.x;
  for (; i < n; i += stride) {
    int d = i % DD;
    int nn = (i / DD) % DKK;
    int h = i / (DD * DKK);
    Wt[((size_t)h * DKK + nn) * DD + d] = f2bf(W[((size_t)h * DD + d) * DKK + nn]);
  }
}

// ---------------- QKV projection: X[b] (SxD) @ W[h] (DxDK) ----------------
__global__ __launch_bounds__(256) void proj_kernel(
    const float* __restrict__ X, const __bf16* __restrict__ Wt,
    __bf16* __restrict__ out, int transpose_out) {
  __shared__ __align__(16) __bf16 xs[128 * 32];  // [row][d]
  __shared__ __align__(16) __bf16 wt[64 * 32];   // [n][d]
  const int tid = threadIdx.x;
  const int tiles_s = SS / 128;
  const int t = blockIdx.x;
  const int srow = (t % tiles_s) * 128;
  const int h = (t / tiles_s) % HH;
  const int b = t / (tiles_s * HH);
  const int wave = tid >> 5, lane = tid & 31;
  const int lrow = lane & 15, g = lane >> 4;
  const int m0 = wave * 16;

  f32x8 acc[4];
#pragma unroll
  for (int i = 0; i < 4; ++i) acc[i] = zero8();

  const float* xb = X + (size_t)b * SS * DD;
  const __bf16* wb = Wt + (size_t)h * DKK * DD;
  for (int d0 = 0; d0 < DD; d0 += 32) {
    // X tile: 512 8-elem units, 16B chunked, f32 -> bf16
#pragma unroll
    for (int r = 0; r < 2; ++r) {
      int u = tid + 256 * r;
      int rr = u >> 2, cc = (u & 3) * 8;
      const float* src = &xb[(size_t)(srow + rr) * DD + d0 + cc];
      f32x4 a0 = *(const f32x4*)src;
      f32x4 a1 = *(const f32x4*)(src + 4);
      Pack8 p;
#pragma unroll
      for (int e = 0; e < 4; ++e) { p.h[e] = f2bf(a0[e]); p.h[4 + e] = f2bf(a1[e]); }
      *(u32x4*)&xs[rr * 32 + cc] = p.q;
    }
    // W tile: straight 16B copies (already transposed)
    {
      int u = tid;
      int nn = u >> 2, cc = (u & 3) * 8;
      *(u32x4*)&wt[nn * 32 + cc] =
          *(const u32x4*)&wb[(size_t)nn * DD + d0 + cc];
    }
    __syncthreads();
    // batch all fragment loads, then back-to-back WMMAs
    Frag a, bf[4];
    a.q[0] = *(const u32x4*)&xs[(m0 + lrow) * 32 + 8 * g];
    a.q[1] = *(const u32x4*)&xs[(m0 + lrow) * 32 + 16 + 8 * g];
#pragma unroll
    for (int nt = 0; nt < 4; ++nt) {
      int n = nt * 16 + lrow;
      bf[nt].q[0] = *(const u32x4*)&wt[n * 32 + 8 * g];
      bf[nt].q[1] = *(const u32x4*)&wt[n * 32 + 16 + 8 * g];
    }
#pragma unroll
    for (int nt = 0; nt < 4; ++nt) acc[nt] = wmma_bf16(a, bf[nt], acc[nt]);
    __syncthreads();
  }
  // branchless strided epilogue
  __bf16* ob = out + ((size_t)b * HH + h) * (size_t)(SS * DKK);
  const size_t rs = transpose_out ? (size_t)1 : (size_t)DKK;
  const size_t cs = transpose_out ? (size_t)SS : (size_t)1;
#pragma unroll
  for (int nt = 0; nt < 4; ++nt)
#pragma unroll
    for (int v = 0; v < 8; ++v) {
      size_t row = (size_t)(srow + m0 + v + 8 * g);
      size_t col = (size_t)(nt * 16 + lrow);
      ob[row * rs + col * cs] = f2bf(acc[nt][v]);
    }
}

// ---------------- flash attention over one (b,h, 128-row q tile) ----------
__global__ __launch_bounds__(256) void flash_kernel(
    const __bf16* __restrict__ Q, const __bf16* __restrict__ K,
    const __bf16* __restrict__ Vt, __bf16* __restrict__ O) {
  __shared__ __align__(16) __bf16 k_lds[2][64 * 64];   // [t][d]
  __shared__ __align__(16) __bf16 v_lds[2][64 * 64];   // [dk][t]
  __shared__ __align__(16) __bf16 p_lds[8 * 16 * 64];  // per-wave [m][t]
  const int tid = threadIdx.x;
  const int tiles_s = SS / 128;
  const int t = blockIdx.x;
  const int srow = (t % tiles_s) * 128;
  const int h = (t / tiles_s) % HH;
  const int b = t / (tiles_s * HH);
  const int wave = tid >> 5, lane = tid & 31;
  const int lrow = lane & 15, g = lane >> 4;
  const int m0 = wave * 16;
  const size_t head = ((size_t)b * HH + h) * SS * DKK;

  Frag qa[2];
  {
    size_t base = head + (size_t)(srow + m0 + lrow) * DKK;
#pragma unroll
    for (int kk = 0; kk < 2; ++kk) {
      qa[kk].q[0] = *(const u32x4*)&Q[base + kk * 32 + 8 * g];
      qa[kk].q[1] = *(const u32x4*)&Q[base + kk * 32 + 16 + 8 * g];
    }
  }
  f32x8 oacc[4];
#pragma unroll
  for (int i = 0; i < 4; ++i) oacc[i] = zero8();
  float m_run[8], l_run[8];
#pragma unroll
  for (int v = 0; v < 8; ++v) { m_run[v] = -3.0e38f; l_run[v] = 0.0f; }

  unsigned koff[2], voff[2];
  koff[0] = (unsigned)(uintptr_t)&k_lds[0][0];
  koff[1] = (unsigned)(uintptr_t)&k_lds[1][0];
  voff[0] = (unsigned)(uintptr_t)&v_lds[0][0];
  voff[1] = (unsigned)(uintptr_t)&v_lds[1][0];
  const __bf16* kbase = K + head;
  const __bf16* vbase = Vt + head;
  const float scale = 0.03125f;  // d_model^-0.5 = 1/32 (faithful to reference)

  // pre-issue tile 0 (double-buffered TDM pipeline)
  if (wave == 0) {
    tdm_load_2d(koff[0], kbase, DKK, SS, DKK, 64, 64);
    tdm_load_2d(voff[0], vbase, SS, DKK, SS, 64, 64);
  }

  for (int t0 = 0; t0 < SS; t0 += 64) {
    const int cur = (t0 >> 6) & 1;
    if (wave == 0) {
      int tn = t0 + 64;
      if (tn < SS) {
        tdm_load_2d(koff[cur ^ 1], kbase + (size_t)tn * DKK, DKK, SS, DKK, 64, 64);
        tdm_load_2d(voff[cur ^ 1], vbase + tn, SS, DKK, SS, 64, 64);
        // next tile's 2 loads outstanding => current tile complete (in-order)
        __builtin_amdgcn_s_wait_tensorcnt(2);
      } else {
        __builtin_amdgcn_s_wait_tensorcnt(0);
      }
    }
    __syncthreads();
    const __bf16* kl = &k_lds[cur][0];
    const __bf16* vl = &v_lds[cur][0];
    // S = Q K^T : per k-step, batch 4 B-frags then 4 WMMAs sharing A
    f32x8 sc[4];
#pragma unroll
    for (int nt = 0; nt < 4; ++nt) sc[nt] = zero8();
#pragma unroll
    for (int kk = 0; kk < 2; ++kk) {
      Frag bk[4];
#pragma unroll
      for (int nt = 0; nt < 4; ++nt) {
        int base = (nt * 16 + lrow) * 64 + kk * 32 + 8 * g;
        bk[nt].q[0] = *(const u32x4*)&kl[base];
        bk[nt].q[1] = *(const u32x4*)&kl[base + 16];
      }
#pragma unroll
      for (int nt = 0; nt < 4; ++nt) sc[nt] = wmma_bf16(qa[kk], bk[nt], sc[nt]);
    }
#pragma unroll
    for (int nt = 0; nt < 4; ++nt)
#pragma unroll
      for (int v = 0; v < 8; ++v) sc[nt][v] *= scale;
    // online softmax: rows span 16-lane groups at fixed accumulator index
    float m_new[8], alpha[8];
#pragma unroll
    for (int v = 0; v < 8; ++v) {
      float mx = fmaxf(fmaxf(sc[0][v], sc[1][v]), fmaxf(sc[2][v], sc[3][v]));
#pragma unroll
      for (int off = 8; off >= 1; off >>= 1)
        mx = fmaxf(mx, __shfl_xor(mx, off, 32));
      float mn = fmaxf(m_run[v], mx);
      alpha[v] = __expf(m_run[v] - mn);
      m_new[v] = mn;
    }
#pragma unroll
    for (int nt = 0; nt < 4; ++nt)
#pragma unroll
      for (int v = 0; v < 8; ++v) sc[nt][v] = __expf(sc[nt][v] - m_new[v]);
#pragma unroll
    for (int v = 0; v < 8; ++v) {
      float s = sc[0][v] + sc[1][v] + sc[2][v] + sc[3][v];
#pragma unroll
      for (int off = 8; off >= 1; off >>= 1) s += __shfl_xor(s, off, 32);
      l_run[v] = l_run[v] * alpha[v] + s;
      m_run[v] = m_new[v];
    }
#pragma unroll
    for (int dn = 0; dn < 4; ++dn)
#pragma unroll
      for (int v = 0; v < 8; ++v) oacc[dn][v] *= alpha[v];
    // C-layout P -> A-layout via per-wave LDS scratch
    __bf16* pw = &p_lds[wave * 16 * 64];
#pragma unroll
    for (int nt = 0; nt < 4; ++nt)
#pragma unroll
      for (int v = 0; v < 8; ++v)
        pw[(v + 8 * g) * 64 + nt * 16 + lrow] = f2bf(sc[nt][v]);
    // O += P V : per k-step, batch A + 4 B-frags, then 4 WMMAs
#pragma unroll
    for (int kk = 0; kk < 2; ++kk) {
      Frag pa, vb[4];
      pa.q[0] = *(const u32x4*)&pw[lrow * 64 + kk * 32 + 8 * g];
      pa.q[1] = *(const u32x4*)&pw[lrow * 64 + kk * 32 + 16 + 8 * g];
#pragma unroll
      for (int dn = 0; dn < 4; ++dn) {
        int base = (dn * 16 + lrow) * 64 + kk * 32 + 8 * g;
        vb[dn].q[0] = *(const u32x4*)&vl[base];
        vb[dn].q[1] = *(const u32x4*)&vl[base + 16];
      }
#pragma unroll
      for (int dn = 0; dn < 4; ++dn) oacc[dn] = wmma_bf16(pa, vb[dn], oacc[dn]);
    }
    __syncthreads();
  }
#pragma unroll
  for (int dn = 0; dn < 4; ++dn)
#pragma unroll
    for (int v = 0; v < 8; ++v) {
      int row = srow + m0 + v + 8 * g;
      int col = h * DKK + dn * 16 + lrow;
      O[((size_t)b * SS + row) * DD + col] = f2bf(oacc[dn][v] / l_run[v]);
    }
}

// ---------------- output projection: O (B*S x D) @ Wo^T ----------------
__global__ __launch_bounds__(256) void outproj_kernel(
    const __bf16* __restrict__ Ob, const __bf16* __restrict__ Wo,
    float* __restrict__ out) {
  __shared__ __align__(16) __bf16 xs[128 * 32];
  __shared__ __align__(16) __bf16 wt[64 * 32];   // Wo rows = output cols
  const int tid = threadIdx.x;
  const int r0 = blockIdx.x * 128;
  const int c0 = blockIdx.y * 64;
  const int wave = tid >> 5, lane = tid & 31;
  const int lrow = lane & 15, g = lane >> 4;
  const int m0 = wave * 16;
  f32x8 acc[4];
#pragma unroll
  for (int i = 0; i < 4; ++i) acc[i] = zero8();
  for (int j0 = 0; j0 < DD; j0 += 32) {
#pragma unroll
    for (int r = 0; r < 2; ++r) {
      int u = tid + 256 * r;
      int rr = u >> 2, cc = (u & 3) * 8;
      *(u32x4*)&xs[rr * 32 + cc] =
          *(const u32x4*)&Ob[(size_t)(r0 + rr) * DD + j0 + cc];
    }
    {
      int u = tid;
      int nn = u >> 2, cc = (u & 3) * 8;
      *(u32x4*)&wt[nn * 32 + cc] =
          *(const u32x4*)&Wo[(size_t)(c0 + nn) * DD + j0 + cc];
    }
    __syncthreads();
    Frag a, bf[4];
    a.q[0] = *(const u32x4*)&xs[(m0 + lrow) * 32 + 8 * g];
    a.q[1] = *(const u32x4*)&xs[(m0 + lrow) * 32 + 16 + 8 * g];
#pragma unroll
    for (int nt = 0; nt < 4; ++nt) {
      int n = nt * 16 + lrow;
      bf[nt].q[0] = *(const u32x4*)&wt[n * 32 + 8 * g];
      bf[nt].q[1] = *(const u32x4*)&wt[n * 32 + 16 + 8 * g];
    }
#pragma unroll
    for (int nt = 0; nt < 4; ++nt) acc[nt] = wmma_bf16(a, bf[nt], acc[nt]);
    __syncthreads();
  }
#pragma unroll
  for (int nt = 0; nt < 4; ++nt)
#pragma unroll
    for (int v = 0; v < 8; ++v) {
      int row = r0 + m0 + v + 8 * g;
      int col = c0 + nt * 16 + lrow;
      out[(size_t)row * DD + col] = acc[nt][v];
    }
}

extern "C" void kernel_launch(void* const* d_in, const int* in_sizes, int n_in,
                              void* d_out, int out_size, void* d_ws,
                              size_t ws_size, hipStream_t stream) {
  (void)in_sizes; (void)n_in; (void)out_size; (void)ws_size;
  const float* q  = (const float*)d_in[0];
  const float* k  = (const float*)d_in[1];
  const float* v  = (const float*)d_in[2];
  const float* Wq = (const float*)d_in[3];
  const float* Wk = (const float*)d_in[4];
  const float* Wv = (const float*)d_in[5];
  const float* Wo = (const float*)d_in[6];
  char* ws = (char*)d_ws;
  const size_t MB = 1024 * 1024;
  __bf16* Wqt = (__bf16*)(ws + 0 * MB);    // [H,DK,D] (transposed)
  __bf16* Wkt = (__bf16*)(ws + 2 * MB);
  __bf16* Wvt = (__bf16*)(ws + 4 * MB);
  __bf16* Wob = (__bf16*)(ws + 6 * MB);    // [D,D] natural
  __bf16* Qb  = (__bf16*)(ws + 8 * MB);    // [B,H,S,DK]
  __bf16* Kb  = (__bf16*)(ws + 40 * MB);   // [B,H,S,DK]
  __bf16* Vtb = (__bf16*)(ws + 72 * MB);   // [B,H,DK,S]
  __bf16* Obf = (__bf16*)(ws + 104 * MB);  // [B,S,D]

  const int nW = HH * DD * DKK;
  cvt_transpose_w<<<1024, 256, 0, stream>>>(Wq, Wqt, nW);
  cvt_transpose_w<<<1024, 256, 0, stream>>>(Wk, Wkt, nW);
  cvt_transpose_w<<<1024, 256, 0, stream>>>(Wv, Wvt, nW);
  cvt_f32_bf16<<<1024, 256, 0, stream>>>(Wo, Wob, DD * DD);

  const int nblk = BB * HH * (SS / 128);   // 2048
  proj_kernel<<<nblk, 256, 0, stream>>>(q, Wqt, Qb, 0);
  proj_kernel<<<nblk, 256, 0, stream>>>(k, Wkt, Kb, 0);
  proj_kernel<<<nblk, 256, 0, stream>>>(v, Wvt, Vtb, 1);

  flash_kernel<<<nblk, 256, 0, stream>>>(Qb, Kb, Vtb, Obf);

  dim3 g2(BB * SS / 128, DD / 64);         // 128 x 16
  outproj_kernel<<<g2, 256, 0, stream>>>(Obf, Wob, (float*)d_out);
}
